// MultiHeadAttention_16372415333143
// MI455X (gfx1250) — matrix operand order repaired and checked by reading, not compiled, not run on hardware
//
#include <hip/hip_runtime.h>

#define D_MODEL 1024
#define N_HEADS 16
#define HEAD_DIM 64
#define BATCH 4
#define SEQ 2048
#define MROWS (BATCH * SEQ)

typedef __attribute__((ext_vector_type(16))) __bf16 v16bf;
typedef __attribute__((ext_vector_type(8)))  float  v8f;
typedef unsigned short u16;

union ABV { v16bf v; uint4 q[2]; };

__device__ __forceinline__ v8f wmma_bf16(v16bf a, v16bf b, v8f c) {
  return __builtin_amdgcn_wmma_f32_16x16x32_bf16(false, a, false, b, (short)0, c, false, false);
}

// hardware v_exp_f32 (base-2 exponential)
__device__ __forceinline__ float fast_exp2(float x) {
  return __builtin_amdgcn_exp2f(x);
}

// fp32 -> bf16 bits, round-to-nearest-even
__device__ __forceinline__ u16 f2bf(float f) {
  unsigned u = __builtin_bit_cast(unsigned, f);
  u = u + 0x7FFFu + ((u >> 16) & 1u);
  return (u16)(u >> 16);
}

// Cross-lane reductions over 16-lane halves. Used ONCE per wave (cold path),
// so the ds_bpermute lowering of __shfl_xor is acceptable here.
__device__ __forceinline__ float rmax16(float v) {
#pragma unroll
  for (int off = 1; off < 16; off <<= 1) v = fmaxf(v, __shfl_xor(v, off, 32));
  return v;
}
__device__ __forceinline__ float rsum16(float v) {
#pragma unroll
  for (int off = 1; off < 16; off <<= 1) v += __shfl_xor(v, off, 32);
  return v;
}

// ---------------- elementwise converts ----------------

__global__ void cvt_bf16_kernel(const float* __restrict__ x, u16* __restrict__ y) {
  int i = blockIdx.x * blockDim.x + threadIdx.x;  // one float4 per thread
  float4 f = reinterpret_cast<const float4*>(x)[i];
  uint2 o;
  o.x = (unsigned)f2bf(f.x) | ((unsigned)f2bf(f.y) << 16);
  o.y = (unsigned)f2bf(f.z) | ((unsigned)f2bf(f.w) << 16);
  reinterpret_cast<uint2*>(y)[i] = o;
}

// W[k][n] fp32 -> Wt[n][k] bf16 (so WMMA B-operand reads are contiguous in k)
__global__ void transpose_w_kernel(const float* __restrict__ W, u16* __restrict__ Wt) {
  int idx = blockIdx.x * blockDim.x + threadIdx.x;  // D*D threads
  int n = idx & (D_MODEL - 1);
  int k = idx >> 10;
  Wt[(size_t)n * D_MODEL + k] = f2bf(W[idx]);
}

// ---------------- WMMA GEMM: C = A(bf16, MROWSxD) @ Wt^T + bias ----------------
// MODE 0: write bf16 head-split [B,H,S,64]   (Q, K)
// MODE 1: write bf16 transposed  [B,H,64,S]  (V)
// MODE 2: write fp32 row-major   [MROWS, D]  (final projection)
template <int MODE>
__global__ __launch_bounds__(128) void gemm_kernel(const u16* __restrict__ A,
                                                   const u16* __restrict__ Bt,
                                                   const float* __restrict__ bias,
                                                   void* __restrict__ dst) {
  const int wave = threadIdx.x >> 5, lane = threadIdx.x & 31;
  const int m = lane & 15, half = lane >> 4;
  const int row0 = blockIdx.x * 64 + wave * 16;
  const int col0 = blockIdx.y * 64;

  v8f acc[4];
#pragma unroll
  for (int g = 0; g < 4; ++g)
#pragma unroll
    for (int r = 0; r < 8; ++r) acc[g][r] = 0.0f;

  const u16* arow = A + (size_t)(row0 + m) * D_MODEL;
  for (int k0 = 0; k0 < D_MODEL; k0 += 32) {
    ABV a;
    a.q[0] = *reinterpret_cast<const uint4*>(arow + k0 + 16 * half);
    a.q[1] = *reinterpret_cast<const uint4*>(arow + k0 + 16 * half + 8);
#pragma unroll
    for (int g = 0; g < 4; ++g) {
      ABV bm;
      const u16* brow = Bt + (size_t)(col0 + g * 16 + m) * D_MODEL + k0 + 16 * half;
      bm.q[0] = *reinterpret_cast<const uint4*>(brow);
      bm.q[1] = *reinterpret_cast<const uint4*>(brow + 8);
      acc[g] = wmma_bf16(a.v, bm.v, acc[g]);
    }
  }

#pragma unroll
  for (int g = 0; g < 4; ++g) {
    const int col = col0 + g * 16 + m;
    const float bb = bias[col];
#pragma unroll
    for (int r = 0; r < 8; ++r) {
      const float val = acc[g][r] + bb;
      const int row = row0 + r + 8 * half;
      if (MODE == 2) {
        reinterpret_cast<float*>(dst)[(size_t)row * D_MODEL + col] = val;
      } else {
        const int b = row >> 11, s = row & (SEQ - 1);
        const int h = col >> 6, hd = col & 63;
        if (MODE == 0)
          reinterpret_cast<u16*>(dst)[(((size_t)(b * N_HEADS + h) * SEQ) + s) * HEAD_DIM + hd] = f2bf(val);
        else
          reinterpret_cast<u16*>(dst)[(((size_t)(b * N_HEADS + h) * HEAD_DIM) + hd) * SEQ + s] = f2bf(val);
      }
    }
  }
}

// ---------------- fused attention: scores + softmax + attn writeback + P@V ----------------

__device__ __forceinline__ v8f score_tile(const u16* __restrict__ Kp, int kv, int m, int half,
                                          v16bf a0, v16bf a1) {
  ABV b0, b1;
  const u16* krow = Kp + (size_t)(kv + m) * HEAD_DIM;
  b0.q[0] = *reinterpret_cast<const uint4*>(krow + 16 * half);
  b0.q[1] = *reinterpret_cast<const uint4*>(krow + 16 * half + 8);
  b1.q[0] = *reinterpret_cast<const uint4*>(krow + 32 + 16 * half);
  b1.q[1] = *reinterpret_cast<const uint4*>(krow + 32 + 16 * half + 8);
  v8f c;
#pragma unroll
  for (int r = 0; r < 8; ++r) c[r] = 0.0f;
  c = wmma_bf16(a0, b0.v, c);
  c = wmma_bf16(a1, b1.v, c);
  return c;
}

__global__ __launch_bounds__(128) void attn_kernel(const u16* __restrict__ Qh,
                                                   const u16* __restrict__ Kh,
                                                   const u16* __restrict__ Vt,
                                                   float* __restrict__ attnw,
                                                   u16* __restrict__ AO) {
  __shared__ u16 Pt[4][16 * 32];  // per-wave bf16 prob tile for operand re-layout

  const int wave = threadIdx.x >> 5, lane = threadIdx.x & 31;
  const int m = lane & 15, half = lane >> 4;
  const int h = blockIdx.y, b = blockIdx.z;
  const int bh = b * N_HEADS + h;
  const int qbase = blockIdx.x * 64 + wave * 16;

  const u16* Qp = Qh + (size_t)bh * SEQ * HEAD_DIM;
  const u16* Kp = Kh + (size_t)bh * SEQ * HEAD_DIM;
  const u16* Vp = Vt + (size_t)bh * HEAD_DIM * SEQ;
  float* attnp = attnw + (size_t)bh * SEQ * SEQ;

  // Q A-operands (hd 0..31 and 32..63) stay in registers for the whole kernel
  ABV a0, a1;
  const u16* qrow = Qp + (size_t)(qbase + m) * HEAD_DIM;
  a0.q[0] = *reinterpret_cast<const uint4*>(qrow + 16 * half);
  a0.q[1] = *reinterpret_cast<const uint4*>(qrow + 16 * half + 8);
  a1.q[0] = *reinterpret_cast<const uint4*>(qrow + 32 + 16 * half);
  a1.q[1] = *reinterpret_cast<const uint4*>(qrow + 32 + 16 * half + 8);

  // log2-domain scale: (1/sqrt(64)) * log2(e)
  const float cl2 = 0.125f * 1.44269504088896f;

  // ---- sweep 1: PER-LANE online max / sum (no cross-lane ops in the loop) ----
  float mrow[8], srow[8];
#pragma unroll
  for (int r = 0; r < 8; ++r) { mrow[r] = -3.0e38f; srow[r] = 0.0f; }

  for (int kv = 0; kv < SEQ; kv += 16) {
    v8f c = score_tile(Kp, kv, m, half, a0.v, a1.v);
#pragma unroll
    for (int r = 0; r < 8; ++r) {
      const float y = c[r] * cl2;
      const float nm = fmaxf(mrow[r], y);
      srow[r] = srow[r] * fast_exp2(mrow[r] - nm) + fast_exp2(y - nm);
      mrow[r] = nm;
    }
  }

  // single cross-lane combine per kernel: global row max + rescaled row sum
  float rinv[8];
#pragma unroll
  for (int r = 0; r < 8; ++r) {
    const float M = rmax16(mrow[r]);
    const float S = rsum16(srow[r] * fast_exp2(mrow[r] - M));
    mrow[r] = M;
    rinv[r] = 1.0f / S;
  }

  // ---- sweep 2: recompute, emit normalized weights, accumulate O += P @ V ----
  v8f acc[4];
#pragma unroll
  for (int g = 0; g < 4; ++g)
#pragma unroll
    for (int r = 0; r < 8; ++r) acc[g][r] = 0.0f;

  u16* myP = Pt[wave];
  for (int kv = 0; kv < SEQ; kv += 32) {
#pragma unroll
    for (int sub = 0; sub < 2; ++sub) {
      const int kv16 = kv + sub * 16;
      v8f c = score_tile(Kp, kv16, m, half, a0.v, a1.v);
#pragma unroll
      for (int r = 0; r < 8; ++r) {
        const float p = fast_exp2(c[r] * cl2 - mrow[r]) * rinv[r];
        const int row = r + 8 * half;
        __builtin_nontemporal_store(p, &attnp[(size_t)(qbase + row) * SEQ + kv16 + m]);
        myP[row * 32 + sub * 16 + m] = f2bf(p);
      }
    }
    __builtin_amdgcn_wave_barrier();  // order cross-lane LDS store -> load within the wave

    ABV ap;
    ap.q[0] = *reinterpret_cast<const uint4*>(myP + m * 32 + 16 * half);
    ap.q[1] = *reinterpret_cast<const uint4*>(myP + m * 32 + 16 * half + 8);
#pragma unroll
    for (int g = 0; g < 4; ++g) {
      ABV bv;
      const u16* vrow = Vp + (size_t)(g * 16 + m) * SEQ + kv + 16 * half;
      bv.q[0] = *reinterpret_cast<const uint4*>(vrow);
      bv.q[1] = *reinterpret_cast<const uint4*>(vrow + 8);
      acc[g] = wmma_bf16(ap.v, bv.v, acc[g]);
    }
    __builtin_amdgcn_wave_barrier();  // keep next iter's stores after this iter's loads
  }

  // write O as bf16 into [B, S, H*hd] for the final projection
#pragma unroll
  for (int g = 0; g < 4; ++g)
#pragma unroll
    for (int r = 0; r < 8; ++r) {
      const int srowi = qbase + r + 8 * half;
      const int col = h * HEAD_DIM + g * 16 + m;
      AO[(size_t)(b * SEQ + srowi) * D_MODEL + col] = f2bf(acc[g][r]);
    }
}

// ---------------- host side ----------------

extern "C" void kernel_launch(void* const* d_in, const int* in_sizes, int n_in,
                              void* d_out, int out_size, void* d_ws, size_t ws_size,
                              hipStream_t stream) {
  (void)in_sizes; (void)n_in; (void)out_size; (void)ws_size;

  const float* q  = (const float*)d_in[0];
  const float* k  = (const float*)d_in[1];
  const float* v  = (const float*)d_in[2];
  const float* wq = (const float*)d_in[3];
  const float* bq = (const float*)d_in[4];
  const float* wk = (const float*)d_in[5];
  const float* bk = (const float*)d_in[6];
  const float* wv = (const float*)d_in[7];
  const float* bv = (const float*)d_in[8];
  const float* wo = (const float*)d_in[9];
  const float* bo = (const float*)d_in[10];

  float* outp  = (float*)d_out;
  float* attnw = outp + (size_t)BATCH * SEQ * D_MODEL;

  char* ws = (char*)d_ws;
  const size_t XB = (size_t)MROWS * D_MODEL * sizeof(u16);    // 16 MB
  const size_t WB = (size_t)D_MODEL * D_MODEL * sizeof(u16);  // 2 MB
  u16* Xq  = (u16*)(ws);
  u16* Xk  = (u16*)(ws + XB);
  u16* Xv  = (u16*)(ws + 2 * XB);
  u16* Wqt = (u16*)(ws + 3 * XB);
  u16* Wkt = (u16*)(ws + 3 * XB + WB);
  u16* Wvt = (u16*)(ws + 3 * XB + 2 * WB);
  u16* Wot = (u16*)(ws + 3 * XB + 3 * WB);
  u16* Qh  = (u16*)(ws + 3 * XB + 4 * WB);
  u16* Kh  = (u16*)(ws + 4 * XB + 4 * WB);
  u16* Vt  = (u16*)(ws + 5 * XB + 4 * WB);
  u16* AO  = (u16*)(ws + 6 * XB + 4 * WB);

  const int cvtBlocks = (MROWS * D_MODEL / 4) / 256;  // float4 per thread
  cvt_bf16_kernel<<<cvtBlocks, 256, 0, stream>>>(q, Xq);
  cvt_bf16_kernel<<<cvtBlocks, 256, 0, stream>>>(k, Xk);
  cvt_bf16_kernel<<<cvtBlocks, 256, 0, stream>>>(v, Xv);

  const int twBlocks = (D_MODEL * D_MODEL) / 256;
  transpose_w_kernel<<<twBlocks, 256, 0, stream>>>(wq, Wqt);
  transpose_w_kernel<<<twBlocks, 256, 0, stream>>>(wk, Wkt);
  transpose_w_kernel<<<twBlocks, 256, 0, stream>>>(wv, Wvt);
  transpose_w_kernel<<<twBlocks, 256, 0, stream>>>(wo, Wot);

  dim3 gg(MROWS / 64, D_MODEL / 64);
  gemm_kernel<0><<<gg, 128, 0, stream>>>(Xq, Wqt, bq, (void*)Qh);
  gemm_kernel<0><<<gg, 128, 0, stream>>>(Xk, Wkt, bk, (void*)Kh);
  gemm_kernel<1><<<gg, 128, 0, stream>>>(Xv, Wvt, bv, (void*)Vt);

  dim3 ga(SEQ / 64, N_HEADS, BATCH);
  attn_kernel<<<ga, 128, 0, stream>>>(Qh, Kh, Vt, attnw, AO);

  gemm_kernel<2><<<gg, 128, 0, stream>>>(AO, Wot, bo, (void*)outp);
}
